// SelfAttentionHead_20134806683803
// MI455X (gfx1250) — compile-verified, matrix operand order checked by reading, not run on hardware
//
#include <hip/hip_runtime.h>
#include <hip/hip_bf16.h>
#include <math.h>

// ---------------------------------------------------------------------------
// Causal single-head self-attention for MI455X (gfx1250, wave32, WMMA).
//   B=256, T=256, D=512, HD=64, fp32 in/out, fp16 compute w/ f32 accumulate.
// v3: - x fp32->fp16 conversion fused into the QKV GEMM (removes a 200MB
//       HBM round trip; x is read exactly once, so staging it bought nothing)
//     - all 12 B fragments of a k-step loaded into distinct registers with a
//       sched_barrier between load group and WMMA group -> staggered partial
//       loadcnt waits instead of wait-0 before every WMMA
//     - attention: load group fenced ahead of the WMMA/softmax group
// ---------------------------------------------------------------------------

typedef _Float16 half_t;
typedef __attribute__((ext_vector_type(8)))  _Float16 v8h;
typedef __attribute__((ext_vector_type(16))) _Float16 v16h;
typedef __attribute__((ext_vector_type(8)))  float    v8f;

#define B_SZ 256
#define T_SZ 256
#define D_SZ 512
#define HD_SZ 64

static __device__ __forceinline__ v8f wmma16x16x32(v16h a, v16h b, v8f c) {
    return __builtin_amdgcn_wmma_f32_16x16x32_f16(
        /*neg_a=*/false, a, /*neg_b=*/false, b,
        /*c_mod=*/(short)0, c, /*reuse_a=*/false, /*reuse_b=*/false);
}

static __device__ __forceinline__ void sched_fence() {
#if __has_builtin(__builtin_amdgcn_sched_barrier)
    __builtin_amdgcn_sched_barrier(0);
#else
    asm volatile("" ::: "memory");
#endif
}

// Per-lane A/B fragment (16-bit, 16x32): lane L holds row/col (L&15),
// halves 0..7 = K[kb8 .. kb8+7], halves 8..15 = K[kb8+16 .. kb8+23],
// kb8 = 8*(L>>4). Two contiguous 16-byte loads from K-major f16 data.
static __device__ __forceinline__ v16h load_frag(const half_t* p) {
    v8h lo = *reinterpret_cast<const v8h*>(p);
    v8h hi = *reinterpret_cast<const v8h*>(p + 16);
    return __builtin_shufflevector(lo, hi, 0,1,2,3,4,5,6,7,8,9,10,11,12,13,14,15);
}

// Same fragment pattern but sourced from fp32 data, converted in registers.
static __device__ __forceinline__ v16h load_cvt_x(const float* p) {
    float4 a0 = *reinterpret_cast<const float4*>(p);
    float4 a1 = *reinterpret_cast<const float4*>(p + 4);
    float4 a2 = *reinterpret_cast<const float4*>(p + 16);
    float4 a3 = *reinterpret_cast<const float4*>(p + 20);
    v16h r;
    r[0]  = (half_t)a0.x; r[1]  = (half_t)a0.y; r[2]  = (half_t)a0.z; r[3]  = (half_t)a0.w;
    r[4]  = (half_t)a1.x; r[5]  = (half_t)a1.y; r[6]  = (half_t)a1.z; r[7]  = (half_t)a1.w;
    r[8]  = (half_t)a2.x; r[9]  = (half_t)a2.y; r[10] = (half_t)a2.z; r[11] = (half_t)a2.w;
    r[12] = (half_t)a3.x; r[13] = (half_t)a3.y; r[14] = (half_t)a3.z; r[15] = (half_t)a3.w;
    return r;
}

static __device__ __forceinline__ v8f zero8() {
    v8f z;
#pragma unroll
    for (int i = 0; i < 8; ++i) z[i] = 0.0f;
    return z;
}

// --------------------------- fp32 -> fp16 convert (W only) ------------------
__global__ void cvt_f32_to_f16_kernel(const float* __restrict__ in,
                                      half_t* __restrict__ out, int n) {
    int i = (blockIdx.x * blockDim.x + threadIdx.x) * 4;
    if (i < n) {
        float4 v = *reinterpret_cast<const float4*>(in + i);
        out[i + 0] = (half_t)v.x;
        out[i + 1] = (half_t)v.y;
        out[i + 2] = (half_t)v.z;
        out[i + 3] = (half_t)v.w;
    }
}

// --------------------------- QKV projection GEMM ----------------------------
// Q[m][n] = sum_k x[m][k] * W[n][k]   (M=B*T, N=64x3, K=512)
// Each wave: one 16-row M tile, 12 accumulator tiles. x is read as fp32 and
// converted to the f16 A fragment in registers (no staging pass).
__global__ void __launch_bounds__(256)
qkv_proj_kernel(const float* __restrict__ x,
                const half_t* __restrict__ wq,
                const half_t* __restrict__ wk,
                const half_t* __restrict__ wv,
                half_t* __restrict__ Qh,   // [B*T][HD], pre-scaled by 0.125
                half_t* __restrict__ Kh,   // [B*T][HD]
                half_t* __restrict__ Vth)  // [B][HD][T]  (transposed V)
{
    const int lane = threadIdx.x & 31;
    const int wave = threadIdx.x >> 5;
    const int tile = blockIdx.x * 8 + wave;   // 16-row tile
    const int row0 = tile * 16;
    const int nl   = lane & 15;
    const int hi   = lane >> 4;
    const int kb8  = hi * 8;

    // Tile i (i = 0..11): output o = i>>2 (Q,K,V), n-tile = i&3.
    auto tile_base = [&](int i) -> const half_t* {
        const half_t* w = (i < 4) ? wq : (i < 8) ? wk : wv;
        return w + (size_t)((i & 3) * 16 + nl) * D_SZ;
    };

    v8f acc[12];
#pragma unroll
    for (int i = 0; i < 12; ++i) acc[i] = zero8();

    const float* xrow = x + (size_t)(row0 + nl) * D_SZ;

    for (int k0 = 0; k0 < D_SZ; k0 += 32) {
        // ---- load group: A (fp32 + cvt) and all 12 B fragments, distinct regs
        v16h a = load_cvt_x(xrow + k0 + kb8);
        v16h bf[12];
#pragma unroll
        for (int i = 0; i < 12; ++i)
            bf[i] = load_frag(tile_base(i) + k0 + kb8);
        sched_fence();
        // ---- compute group: 12 WMMAs drain the loads with staggered waits
#pragma unroll
        for (int i = 0; i < 12; ++i)
            acc[i] = wmma16x16x32(a, bf[i], acc[i]);
        sched_fence();
    }

    // C/D layout: lane holds (col n = nl, rows m = r + 8*hi), r = 0..7.
    const int bidx = row0 >> 8;       // row0 / T
    const int t0   = row0 & (T_SZ - 1);
#pragma unroll
    for (int n = 0; n < 4; ++n) {
#pragma unroll
        for (int r = 0; r < 8; ++r) {
            const int m = r + 8 * hi;
            Qh[(size_t)(row0 + m) * HD_SZ + n * 16 + nl] = (half_t)(acc[n][r] * 0.125f);
            Kh[(size_t)(row0 + m) * HD_SZ + n * 16 + nl] = (half_t)acc[4 + n][r];
        }
        // V^T: per lane the 8 accum rows are consecutive t -> one b128 store.
        v8h vp;
#pragma unroll
        for (int r = 0; r < 8; ++r) vp[r] = (half_t)acc[8 + n][r];
        half_t* vdst = Vth + ((size_t)bidx * HD_SZ + n * 16 + nl) * T_SZ + t0 + 8 * hi;
        *reinterpret_cast<v8h*>(vdst) = vp;
    }
}

// ------------------------------ flash attention -----------------------------
// Per wave: one 16-query tile. S^T = K Q^T puts the softmax row (fixed query)
// along a lane's 8 VGPRs; P^T re-packs lane-locally into the B fragment of
// O^T = V^T P^T (no shuffles / LDS transpose).
__global__ void __launch_bounds__(256)
attn_kernel(const half_t* __restrict__ Qh,
            const half_t* __restrict__ Kh,
            const half_t* __restrict__ Vth,
            float* __restrict__ out)   // [B][T][HD]
{
    const int lane = threadIdx.x & 31;
    const int wave = threadIdx.x >> 5;
    const int tidx = blockIdx.x * 8 + wave;      // global query-tile index
    const int b    = tidx >> 4;                   // T/16 = 16 tiles per batch
    const int q0   = (tidx & 15) << 4;
    const int nl   = lane & 15;
    const int hi   = lane >> 4;
    const int kb8  = hi * 8;

    // Q^T B fragments, resident across the whole key loop.
    const half_t* qrow = Qh + ((size_t)b * T_SZ + q0 + nl) * HD_SZ;
    const v16h qf0 = load_frag(qrow + kb8);
    const v16h qf1 = load_frag(qrow + 32 + kb8);

    const half_t* kbase = Kh + (size_t)b * T_SZ * HD_SZ;
    const half_t* vbase = Vth + (size_t)b * HD_SZ * T_SZ;

    float mrun = -INFINITY;
    float lrun = 0.0f;
    v8f ot[4];
#pragma unroll
    for (int mt = 0; mt < 4; ++mt) ot[mt] = zero8();

    const int nch = (q0 >> 5) + 1;               // 32-key chunks (causal bound)
    for (int j = 0; j < nch; ++j) {
        const int kb = j * 32;

        // ---- load group: 4 K fragments + 4 V fragments, distinct regs.
        const half_t* krow0 = kbase + (size_t)(kb + nl) * HD_SZ;
        const half_t* krow1 = kbase + (size_t)(kb + 16 + nl) * HD_SZ;
        const v16h kf00 = load_frag(krow0 + kb8);
        const v16h kf01 = load_frag(krow0 + 32 + kb8);
        const v16h kf10 = load_frag(krow1 + kb8);
        const v16h kf11 = load_frag(krow1 + 32 + kb8);
        const v16h vf0 = load_frag(vbase + (size_t)(0 * 16 + nl) * T_SZ + kb + kb8);
        const v16h vf1 = load_frag(vbase + (size_t)(1 * 16 + nl) * T_SZ + kb + kb8);
        const v16h vf2 = load_frag(vbase + (size_t)(2 * 16 + nl) * T_SZ + kb + kb8);
        const v16h vf3 = load_frag(vbase + (size_t)(3 * 16 + nl) * T_SZ + kb + kb8);

        // ---- prefetch next chunk (global_prefetch_b8) during softmax math.
        if (j + 1 < nch) {
            const int kn = kb + 32;
            __builtin_prefetch(kbase + (size_t)(kn + nl) * HD_SZ + kb8, 0, 0);
            __builtin_prefetch(kbase + (size_t)(kn + 16 + nl) * HD_SZ + kb8, 0, 0);
            __builtin_prefetch(vbase + (size_t)(0 * 16 + nl) * T_SZ + kn + kb8, 0, 0);
            __builtin_prefetch(vbase + (size_t)(2 * 16 + nl) * T_SZ + kn + kb8, 0, 0);
        }
        sched_fence();

        v8f s0 = zero8(), s1 = zero8();
        s0 = wmma16x16x32(kf00, qf0, s0);
        s0 = wmma16x16x32(kf01, qf1, s0);
        s1 = wmma16x16x32(kf10, qf0, s1);
        s1 = wmma16x16x32(kf11, qf1, s1);

        // ---- causal mask (only the diagonal chunk can violate key <= query)
        if (j == nch - 1) {
            const int qg  = q0 + nl;
            const int kg0 = kb + 8 * hi;
#pragma unroll
            for (int r = 0; r < 8; ++r) {
                if (kg0 + r > qg)      s0[r] = -INFINITY;
                if (kg0 + 16 + r > qg) s1[r] = -INFINITY;
            }
        }

        // ---- online softmax: row = query = lane&15; partner = lane^16.
        float cmax = s0[0];
#pragma unroll
        for (int r = 1; r < 8; ++r) cmax = fmaxf(cmax, s0[r]);
#pragma unroll
        for (int r = 0; r < 8; ++r) cmax = fmaxf(cmax, s1[r]);
        cmax = fmaxf(cmax, __shfl_xor(cmax, 16, 32));
        const float mnew = fmaxf(mrun, cmax);
        const float corr = __expf(mrun - mnew);

        v16h pt;                                  // P^T B fragment, lane-local
        float psum = 0.0f;
#pragma unroll
        for (int r = 0; r < 8; ++r) {
            float p = __expf(s0[r] - mnew);
            psum += p;
            pt[r] = (half_t)p;
        }
#pragma unroll
        for (int r = 0; r < 8; ++r) {
            float p = __expf(s1[r] - mnew);
            psum += p;
            pt[8 + r] = (half_t)p;
        }
        psum += __shfl_xor(psum, 16, 32);
        lrun = lrun * corr + psum;
        mrun = mnew;

#pragma unroll
        for (int mt = 0; mt < 4; ++mt)
#pragma unroll
            for (int r = 0; r < 8; ++r) ot[mt][r] *= corr;

        // ---- O^T += V^T(16h x 32k) * P^T(32k x 16q), 4 h-tiles
        ot[0] = wmma16x16x32(vf0, pt, ot[0]);
        ot[1] = wmma16x16x32(vf1, pt, ot[1]);
        ot[2] = wmma16x16x32(vf2, pt, ot[2]);
        ot[3] = wmma16x16x32(vf3, pt, ot[3]);
    }

    // ---- writeback: O^T element (h = mt*16 + r + 8*hi, q = q0 + nl).
    // Per lane the 8 accum rows are consecutive h -> two float4 stores.
    const float inv = 1.0f / lrun;
    float* orow = out + ((size_t)b * T_SZ + q0 + nl) * HD_SZ;
#pragma unroll
    for (int mt = 0; mt < 4; ++mt) {
        float4 lo4 = make_float4(ot[mt][0] * inv, ot[mt][1] * inv,
                                 ot[mt][2] * inv, ot[mt][3] * inv);
        float4 hi4 = make_float4(ot[mt][4] * inv, ot[mt][5] * inv,
                                 ot[mt][6] * inv, ot[mt][7] * inv);
        float* p = orow + mt * 16 + 8 * hi;
        *reinterpret_cast<float4*>(p)     = lo4;
        *reinterpret_cast<float4*>(p + 4) = hi4;
    }
}

// ------------------------------------------------------------------------
extern "C" void kernel_launch(void* const* d_in, const int* in_sizes, int n_in,
                              void* d_out, int out_size, void* d_ws, size_t ws_size,
                              hipStream_t stream) {
    const float* x  = (const float*)d_in[0];   // [B,T,D]
    const float* Wq = (const float*)d_in[1];   // [HD,D]
    const float* Wk = (const float*)d_in[2];
    const float* Wv = (const float*)d_in[3];
    float* out = (float*)d_out;                // [B,T,HD]

    // Workspace layout (f16), ~26 MB total (x no longer staged).
    char* ws = (char*)d_ws;
    const size_t n_w  = (size_t)HD_SZ * D_SZ;             // 32768
    const size_t n_qk = (size_t)B_SZ * T_SZ * HD_SZ;      // 4.19M
    half_t* wqh = (half_t*)ws;
    half_t* wkh = wqh + n_w;
    half_t* wvh = wkh + n_w;
    half_t* Qh  = wvh + n_w;
    half_t* Kh  = Qh + n_qk;
    half_t* Vth = Kh + n_qk;
    (void)ws_size; (void)in_sizes; (void)n_in; (void)out_size;

    // 1) weight fp32 -> fp16 (tiny)
    cvt_f32_to_f16_kernel<<<(int)(n_w / 1024), 256, 0, stream>>>(Wq, wqh, (int)n_w);
    cvt_f32_to_f16_kernel<<<(int)(n_w / 1024), 256, 0, stream>>>(Wk, wkh, (int)n_w);
    cvt_f32_to_f16_kernel<<<(int)(n_w / 1024), 256, 0, stream>>>(Wv, wvh, (int)n_w);

    // 2) QKV projection (fused x conversion): 4096 16-row tiles, 8 waves/block
    qkv_proj_kernel<<<512, 256, 0, stream>>>(x, wqh, wkh, wvh, Qh, Kh, Vth);

    // 3) flash attention: 4096 query tiles, 8 waves/block
    attn_kernel<<<512, 256, 0, stream>>>(Qh, Kh, Vth, out);
}
